// TemporalBlockLayer_84215718739981
// MI455X (gfx1250) — matrix operand order repaired and checked by reading, not compiled, not run on hardware
//
#include <hip/hip_runtime.h>
#include <stdint.h>

// TemporalBlockLayer for MI455X (gfx1250): bf16 WMMA GEMM pipeline, v2.
// B=8, T=2048, C=512, H=4, K=D=128, conv KS=4 DIL=4, band=3.
//
// v2 changes vs v1:
//  - All weight matrices pre-packed into WMMA B-fragment layout
//    ([ntile][ktile][lane][16 halves]); GEMM loops read B fragments with two
//    coalesced global_load_b128 per lane from L2 (192 MB -> weights resident).
//    Removes the transposed LDS staging (the VALU/u16-load hotspot in v1).
//  - A tiles staged global->LDS with CDNA5 async copies
//    (global_load_async_to_lds_b128 + s_wait_asynccnt), double-buffered so the
//    async stage of tile k+1 overlaps the WMMAs of tile k.

#define TDIM 2048
#define CDIM 512
#define PADR 12          // (KS-1)*DIL
#define TPAD (TDIM + PADR)

typedef __attribute__((ext_vector_type(16))) __bf16 v16bf;
typedef __attribute__((ext_vector_type(8)))  float  v8f;

struct U128 { unsigned int x, y, z, w; };
struct US4  { unsigned short x, y, z, w; };
union  Frag { v16bf v; U128 q[2]; };

__device__ __forceinline__ unsigned short f2bf(float f) {
  unsigned int u = __float_as_uint(f);
  u += 0x7FFFu + ((u >> 16) & 1u);         // round-to-nearest-even
  return (unsigned short)(u >> 16);
}

__device__ __forceinline__ v8f wmma_bf16(v16bf a, v16bf b, v8f c) {
  // (neg_a, A, neg_b, B, c_mod, C, reuse_a, reuse_b)
  return __builtin_amdgcn_wmma_f32_16x16x32_bf16(false, a, false, b,
                                                 (short)0, c, false, false);
}

// ---- async A staging: 64 rows x 32 halves -> LDS, 256 thr x 16B ----
__device__ __forceinline__ void stageA_async(unsigned short* __restrict__ dst,
                                             const unsigned short* __restrict__ src,
                                             int lda, int kk, int tid) {
  int row = tid >> 2, chunk = (tid & 3) * 8;
  const unsigned short* s = src + (size_t)row * lda + kk + chunk;
  // low 32 bits of a generic pointer into __shared__ = LDS byte offset
  unsigned lds = (unsigned)(size_t)(dst + row * 32 + chunk);
  unsigned long long ga = (unsigned long long)(size_t)s;
  __builtin_prefetch((const void*)(s + 32), 0, 1);   // hint next k-tile to L2
  asm volatile("global_load_async_to_lds_b128 %0, %1, off"
               :: "v"(lds), "v"(ga) : "memory");
}

__device__ __forceinline__ void wait_async0() {
  asm volatile("s_wait_asynccnt 0x0" ::: "memory");
}

// ---- A fragment from LDS (16-bit A 16x32, contiguous k = 16*(lane/16) + s, same as B packing) ----
__device__ __forceinline__ v16bf loadA(const unsigned short* tile, int lane) {
  int m = lane & 15, kg = lane >> 4;
  const unsigned short* r = tile + m * 32 + kg * 16;
  Frag f;
  f.q[0] = *(const U128*)r;          // K = kg*16 .. kg*16+7
  f.q[1] = *(const U128*)(r + 8);    // K = kg*16+8 .. kg*16+15
  return f.v;
}

// ---- B fragment straight from pre-packed global (frag layout) ----
__device__ __forceinline__ v16bf loadBfrag(const unsigned short* bp) {
  Frag f;                            // 32 contiguous bytes per lane
  f.q[0] = *(const U128*)bp;
  f.q[1] = *(const U128*)(bp + 8);
  return f.v;
}

// ================= generic bf16 WMMA GEMM =================
// Block 256 thr (8 waves), block tile 64x64, BK=32. LDS holds only A
// (2 x 4 KB, double buffered via async copies). B fragments from L2.
// pad>0 remaps output row m -> b*(Trows+pad)+pad+t (causal-conv padding).
__global__ __launch_bounds__(256)
void gemm_bf16_wmma_kernel(const unsigned short* __restrict__ A, int lda,
                           const unsigned short* __restrict__ Brep,
                           const float* __restrict__ bias,
                           float* __restrict__ Cf,
                           unsigned short* __restrict__ Cb,
                           int ldc, int Kd, int Trows, int pad) {
  __shared__ unsigned short As[2][64 * 32];
  const int tid = threadIdx.x, lane = tid & 31, wave = tid >> 5;
  const int wm = wave >> 2, wn = wave & 3;
  const int nb = blockIdx.x * 64, mb = blockIdx.y * 64;
  const int ktiles = Kd >> 5;

  v8f acc0 = {0,0,0,0,0,0,0,0}, acc1 = {0,0,0,0,0,0,0,0};
  const unsigned short* Ab = A + (size_t)mb * lda;
  // this wave's column tile in fragment layout: [nt][kt][lane][16]
  const unsigned short* Bw =
      Brep + ((size_t)(nb / 16 + wn) * ktiles * 32 + lane) * 16;

  stageA_async(As[0], Ab, lda, 0, tid);
  wait_async0();
  __syncthreads();
  int buf = 0;
  for (int kt = 0; kt < ktiles; ++kt) {
    const bool more = (kt + 1) < ktiles;
    if (more) stageA_async(As[buf ^ 1], Ab, lda, (kt + 1) * 32, tid);
    v16bf bf = loadBfrag(Bw + (size_t)kt * 512);   // 32 lanes * 16 halves
    v16bf a0 = loadA(&As[buf][(wm * 32) * 32], lane);
    v16bf a1 = loadA(&As[buf][(wm * 32 + 16) * 32], lane);
    acc0 = wmma_bf16(a0, bf, acc0);
    acc1 = wmma_bf16(a1, bf, acc1);
    if (more) wait_async0();
    __syncthreads();
    buf ^= 1;
  }

  const int n = nb + wn * 16 + (lane & 15);
  const float bv = bias ? bias[n] : 0.0f;
  const int mh = (lane >> 4) * 8;
#pragma unroll
  for (int r = 0; r < 8; ++r) {
    int m0 = mb + wm * 32 + mh + r;
    int m1 = m0 + 16;
    float v0 = acc0[r] + bv, v1 = acc1[r] + bv;
    size_t row0, row1;
    if (pad == 0) { row0 = (size_t)m0; row1 = (size_t)m1; }
    else {
      int b0 = m0 / Trows, b1 = m1 / Trows;
      row0 = (size_t)b0 * (Trows + pad) + pad + (m0 - b0 * Trows);
      row1 = (size_t)b1 * (Trows + pad) + pad + (m1 - b1 * Trows);
    }
    if (Cf) { Cf[row0 * ldc + n] = v0; Cf[row1 * ldc + n] = v1; }
    if (Cb) { Cb[row0 * ldc + n] = f2bf(v0); Cb[row1 * ldc + n] = f2bf(v1); }
  }
}

// ================= band attention (softmax over <=7 neighbors) =============
// One wave32 per (b,h,t). Lane holds 4 of 128 channels; cross-lane reduce for
// q.k dots; softmax in registers; writes cat[b*T+t][h*128+d] as bf16.
__global__ __launch_bounds__(256)
void band_attn_kernel(const float* __restrict__ qkv,      // [16384][1536]
                      unsigned short* __restrict__ cat) { // [16384][512] bf16
  const int lane = threadIdx.x & 31;
  const int wid  = (blockIdx.x * blockDim.x + threadIdx.x) >> 5; // 0..65535
  const int h = wid & 3;
  const int t = (wid >> 2) & (TDIM - 1);
  const int b = wid >> 13;

  const size_t rowq = (size_t)(b * TDIM + t) * 1536 + h * 128;
  float4 q4 = *(const float4*)(qkv + rowq + lane * 4);

  const int s0 = (t - 3 < 0) ? 0 : t - 3;
  const int s1 = (t + 3 > TDIM - 1) ? TDIM - 1 : t + 3;
  const int cnt = s1 - s0 + 1;
  float sc[7];
  for (int s = s0; s <= s1; ++s) {
    const float* kp = qkv + (size_t)(b * TDIM + s) * 1536 + 512 + h * 128;
    float4 k4 = *(const float4*)(kp + lane * 4);
    float d = q4.x * k4.x + q4.y * k4.y + q4.z * k4.z + q4.w * k4.w;
    for (int off = 16; off > 0; off >>= 1) d += __shfl_xor(d, off, 32);
    sc[s - s0] = d * 0.08838834764831845f;   // 1/sqrt(128)
  }
  float mx = sc[0];
  for (int i = 1; i < cnt; ++i) mx = fmaxf(mx, sc[i]);
  float sum = 0.f;
  for (int i = 0; i < cnt; ++i) { sc[i] = __expf(sc[i] - mx); sum += sc[i]; }
  const float inv = 1.f / sum;

  float4 acc = make_float4(0.f, 0.f, 0.f, 0.f);
  for (int i = 0; i < cnt; ++i) {
    const float* vp = qkv + (size_t)(b * TDIM + s0 + i) * 1536 + 1024 + h * 128;
    float4 v4 = *(const float4*)(vp + lane * 4);
    float w = sc[i] * inv;
    acc.x += w * v4.x; acc.y += w * v4.y; acc.z += w * v4.z; acc.w += w * v4.w;
  }
  US4 o = { f2bf(acc.x), f2bf(acc.y), f2bf(acc.z), f2bf(acc.w) };
  *(US4*)(cat + (size_t)(b * TDIM + t) * CDIM + h * 128 + lane * 4) = o;
}

// ================= dilated conv as WMMA GEMM + fused epilogue ==============
// out[b,t,:] = leaky( relu( sum_j outp[b, t+4j (padded)] @ Wc[j] + bc ) + x )
__global__ __launch_bounds__(256)
void conv_fused_wmma_kernel(const unsigned short* __restrict__ Ap,   // [B][TPAD][C] bf16
                            const unsigned short* __restrict__ Wcr,  // 4 x frag layout
                            const float* __restrict__ bc,
                            const float* __restrict__ xres,
                            float* __restrict__ out) {
  __shared__ unsigned short As[2][64 * 32];
  const int tid = threadIdx.x, lane = tid & 31, wave = tid >> 5;
  const int wm = wave >> 2, wn = wave & 3;
  const int nb = blockIdx.x * 64, mb = blockIdx.y * 64;
  const int b = mb >> 11, t0 = mb & (TDIM - 1);   // 64 | 2048: no batch straddle

  v8f acc0 = {0,0,0,0,0,0,0,0}, acc1 = {0,0,0,0,0,0,0,0};
  // per tap: frag base for this wave's column tile (ktiles = 16 per tap)
  const size_t bwoff = ((size_t)(nb / 16 + wn) * 16 * 32 + lane) * 16;

  stageA_async(As[0], Ap + ((size_t)b * TPAD + t0) * CDIM, CDIM, 0, tid);
  wait_async0();
  __syncthreads();
  int buf = 0;
  for (int s = 0; s < 64; ++s) {                  // 4 taps x 16 k-steps
    const bool more = (s + 1) < 64;
    if (more) {
      int jn = (s + 1) >> 4, kkn = ((s + 1) & 15) * 32;
      stageA_async(As[buf ^ 1], Ap + ((size_t)b * TPAD + t0 + 4 * jn) * CDIM,
                   CDIM, kkn, tid);
    }
    int j = s >> 4, kt = s & 15;
    v16bf bf = loadBfrag(Wcr + ((size_t)j << 18) + bwoff + (size_t)kt * 512);
    v16bf a0 = loadA(&As[buf][(wm * 32) * 32], lane);
    v16bf a1 = loadA(&As[buf][(wm * 32 + 16) * 32], lane);
    acc0 = wmma_bf16(a0, bf, acc0);
    acc1 = wmma_bf16(a1, bf, acc1);
    if (more) wait_async0();
    __syncthreads();
    buf ^= 1;
  }

  const int n = nb + wn * 16 + (lane & 15);
  const float bcv = bc[n];
  const int mh = (lane >> 4) * 8;
#pragma unroll
  for (int r = 0; r < 8; ++r) {
    int m0 = mb + wm * 32 + mh + r;
    int m1 = m0 + 16;
    float v0 = fmaxf(acc0[r] + bcv, 0.0f) + xres[(size_t)m0 * CDIM + n];
    float v1 = fmaxf(acc1[r] + bcv, 0.0f) + xres[(size_t)m1 * CDIM + n];
    out[(size_t)m0 * CDIM + n] = (v0 >= 0.f) ? v0 : 0.2f * v0;
    out[(size_t)m1 * CDIM + n] = (v1 >= 0.f) ? v1 : 0.2f * v1;
  }
}

// ================= pack / convert kernels =================
__global__ void cvt4_f32_bf16_kernel(const float* __restrict__ in,
                                     unsigned short* __restrict__ o, int n4) {
  int i = blockIdx.x * 256 + threadIdx.x;
  if (i < n4) {
    float4 v = ((const float4*)in)[i];
    US4 r = { f2bf(v.x), f2bf(v.y), f2bf(v.z), f2bf(v.w) };
    ((US4*)o)[i] = r;
  }
}

// pack a row-major f32 [Kd][N] into B fragment layout [nt][kt][lane][16]
__global__ void pack_frag_kernel(const float* __restrict__ src,
                                 unsigned short* __restrict__ rep,
                                 int N, int Kd) {
  int i = blockIdx.x * 256 + threadIdx.x;   // (N/16)*(Kd/32)*32 threads, exact
  int lane = i & 31;
  int kt = (i >> 5) % (Kd >> 5);
  int nt = (i >> 5) / (Kd >> 5);
  int nl = lane & 15, kh = lane >> 4;
  int nn = nt * 16 + nl;
  unsigned short* dst = rep + (size_t)i * 16;
#pragma unroll
  for (int e = 0; e < 16; ++e) {
    int k = kt * 32 + kh * 16 + e;
    dst[e] = f2bf(src[(size_t)k * N + nn]);
  }
}

// pack {Wq,Wk,Wv}[H,C,128] -> W1 frag layout for [Kd=512][N=1536]
__global__ void pack_w1_frag_kernel(const float* __restrict__ Wq,
                                    const float* __restrict__ Wk,
                                    const float* __restrict__ Wv,
                                    unsigned short* __restrict__ rep) {
  int i = blockIdx.x * 256 + threadIdx.x;   // 96*16*32 = 49152 threads, exact
  int lane = i & 31;
  int kt = (i >> 5) & 15;                   // Kd/32 = 16
  int nt = i >> 9;                          // 0..95
  int nl = lane & 15, kh = lane >> 4;
  int nn = nt * 16 + nl;
  unsigned short* dst = rep + (size_t)i * 16;
#pragma unroll
  for (int e = 0; e < 16; ++e) {
    int c = kt * 32 + kh * 16 + e;          // input channel (K of the GEMM)
    float w;
    if (nn < 512)       { int h = nn >> 7,          k = nn & 127; w = Wq[((size_t)h * CDIM + c) * 128 + k]; }
    else if (nn < 1024) { int h = (nn - 512) >> 7,  k = nn & 127; w = Wk[((size_t)h * CDIM + c) * 128 + k]; }
    else                { int h = (nn - 1024) >> 7, k = nn & 127; w = Wv[((size_t)h * CDIM + c) * 128 + k]; }
    dst[e] = f2bf(w);
  }
}

__global__ void pack_bias1_kernel(const float* __restrict__ bq,
                                  const float* __restrict__ bk,
                                  const float* __restrict__ bv,
                                  float* __restrict__ b1) {
  int i = blockIdx.x * 256 + threadIdx.x;
  if (i < 1536) b1[i] = (i < 512) ? bq[i] : (i < 1024 ? bk[i - 512] : bv[i - 1024]);
}

__global__ void zero_pad_kernel(unsigned short* __restrict__ outp) {
  int i = blockIdx.x * 256 + threadIdx.x;   // 8*12*512 threads, exact
  int b = i / (PADR * CDIM);
  int r = i - b * (PADR * CDIM);
  outp[(size_t)b * TPAD * CDIM + r] = 0;
}

// ================= launcher =================
extern "C" void kernel_launch(void* const* d_in, const int* in_sizes, int n_in,
                              void* d_out, int out_size, void* d_ws, size_t ws_size,
                              hipStream_t stream) {
  const float* x  = (const float*)d_in[0];
  const float* Wq = (const float*)d_in[1];
  const float* bq = (const float*)d_in[2];
  const float* Wk = (const float*)d_in[3];
  const float* bk = (const float*)d_in[4];
  const float* Wv = (const float*)d_in[5];
  const float* bv = (const float*)d_in[6];
  const float* Wo = (const float*)d_in[7];
  const float* bo = (const float*)d_in[8];
  const float* Wc = (const float*)d_in[9];
  const float* bc = (const float*)d_in[10];
  float* out = (float*)d_out;

  char* ws = (char*)d_ws;                                   // ~155.3 MB used
  unsigned short* xb   = (unsigned short*)(ws);             // 16 MB  x bf16
  unsigned short* W1r  = (unsigned short*)(ws + 16777216);  // 1.5 MB qkv W frag
  float*          b1   = (float*)         (ws + 18350080);  // 6 KB   qkv bias
  unsigned short* Wor  = (unsigned short*)(ws + 18356224);  // 0.5 MB Wo frag
  unsigned short* Wcr  = (unsigned short*)(ws + 18880512);  // 2 MB   Wc frag x4
  float*          qkv  = (float*)         (ws + 20977664);  // 96 MB  q|k|v f32
  unsigned short* cat  = (unsigned short*)(ws + 121640960); // 16 MB  heads bf16
  unsigned short* outp = (unsigned short*)(ws + 138418176); // 16 MB  padded out

  dim3 blk(256);
  cvt4_f32_bf16_kernel<<<8192, blk, 0, stream>>>(x, xb, (16384 * 512) / 4);
  pack_w1_frag_kernel<<<192, blk, 0, stream>>>(Wq, Wk, Wv, W1r);
  pack_bias1_kernel<<<6, blk, 0, stream>>>(bq, bk, bv, b1);
  pack_frag_kernel<<<64, blk, 0, stream>>>(Wo, Wor, 512, 512);
  for (int j = 0; j < 4; ++j)
    pack_frag_kernel<<<64, blk, 0, stream>>>(Wc + (size_t)j * 262144,
                                             Wcr + (size_t)j * 262144, 512, 512);
  zero_pad_kernel<<<192, blk, 0, stream>>>(outp);

  // qkv = x @ W1 + bias1  -> f32 [16384,1536]
  gemm_bf16_wmma_kernel<<<dim3(24, 256), blk, 0, stream>>>(
      xb, 512, W1r, b1, qkv, nullptr, 1536, 512, 0, 0);
  // band softmax + weighted V -> cat bf16 [16384,512]
  band_attn_kernel<<<8192, blk, 0, stream>>>(qkv, cat);
  // out = cat @ Wo + bo -> bf16 into padded [B][T+12][512]
  gemm_bf16_wmma_kernel<<<dim3(8, 256), blk, 0, stream>>>(
      cat, 512, Wor, bo, nullptr, outp, 512, 512, TDIM, PADR);
  // dilated conv (4 taps) + bias + relu + residual + leaky -> d_out f32
  conv_fused_wmma_kernel<<<dim3(8, 256), blk, 0, stream>>>(outp, Wcr, bc, x, out);
}
